// SimpleAttention_21225728377093
// MI455X (gfx1250) — matrix operand order repaired and checked by reading, not compiled, not run on hardware
//
#include <hip/hip_runtime.h>
#include <hip/hip_bf16.h>
#include <math.h>

typedef __attribute__((ext_vector_type(16))) __bf16 v16bf;
typedef __attribute__((ext_vector_type(8)))  __bf16 v8bf;
typedef __attribute__((ext_vector_type(8)))  float  v8f;
typedef __attribute__((ext_vector_type(4)))  unsigned int v4ui;
typedef __attribute__((ext_vector_type(8)))  int v8i;
typedef __attribute__((ext_vector_type(4)))  int v4i;

constexpr int Dm = 512;   // feature dim
constexpr int Bm = 64;    // batch
constexpr int Sm = 2048;  // seq

__device__ __forceinline__ unsigned short f32_bf16(float f) {
  unsigned u = __float_as_uint(f);
  u += 0x7FFFu + ((u >> 16) & 1u);   // round-to-nearest-even
  return (unsigned short)(u >> 16);
}

// Issue one TDM DMA: 2-D tile (tile_dim0 x tile_dim1) of bf16 (2B) elements,
// row stride `stride0` elements, from global `gptr` into LDS byte offset `ldsOff`.
__device__ __forceinline__ void tdm_load_2d_bf16(const void* gptr, unsigned ldsOff,
                                                 unsigned tileD0, unsigned tileD1,
                                                 unsigned tensD0, unsigned tensD1,
                                                 unsigned stride0) {
  unsigned long long ga = (unsigned long long)(uintptr_t)gptr;
  v4ui g0;
  g0[0] = 1u;                                    // count=1, user mode
  g0[1] = ldsOff;                                // lds_addr (bytes)
  g0[2] = (unsigned)(ga & 0xFFFFFFFFu);          // global_addr[31:0]
  g0[3] = (unsigned)((ga >> 32) & 0x1FFFFFFu)    // global_addr[56:32]
        | (2u << 30);                            // type=2 ("image")
  v8i g1;
  g1[0] = (int)(1u << 16);                       // data_size=1 -> 2 bytes
  g1[1] = (int)(tensD0 << 16);                   // tensor_dim0[15:0]
  g1[2] = (int)((tensD0 >> 16) | (tensD1 << 16));// tensor_dim0 hi | tensor_dim1 lo
  g1[3] = (int)((tensD1 >> 16) | (tileD0 << 16));// tensor_dim1 hi | tile_dim0
  g1[4] = (int)(tileD1 & 0xFFFFu);               // tile_dim1 | tile_dim2=0
  g1[5] = (int)stride0;                          // tensor_dim0_stride[31:0]
  g1[6] = 0;                                     // stride0 hi | stride1 lo (unused, 2D)
  g1[7] = 0;
  v4i z4 = {0, 0, 0, 0};
#if __has_include(<hip/amd_detail/amd_gfx1250_TDM.h>)
  v8i z8 = {0, 0, 0, 0, 0, 0, 0, 0};
  __builtin_amdgcn_tensor_load_to_lds(g0, g1, z4, z4, z8, 0);   // clang-23 / therock form
#else
  __builtin_amdgcn_tensor_load_to_lds(g0, g1, z4, z4, 0);       // ROCm 7.2 form
#endif
}

// ---------------- weight prep: bf16 convert Wq/Wk ----------------
__global__ void convert_w_kernel(const float* __restrict__ Wq,
                                 const float* __restrict__ Wk,
                                 unsigned short* __restrict__ Wqb,
                                 unsigned short* __restrict__ Wkb) {
  int i = blockIdx.x * blockDim.x + threadIdx.x;
  if (i < Dm * Dm) {
    Wqb[i] = f32_bf16(Wq[i]);
    Wkb[i] = f32_bf16(Wk[i]);
  }
}

// ---------------- Wv column sums (Σ_d Wv[d,:]) and Σ bv ----------------
__global__ void wv_colsum_kernel(const float* __restrict__ Wv,
                                 const float* __restrict__ bv,
                                 float* __restrict__ wvsum,
                                 float* __restrict__ bvsum) {
  int k = blockIdx.x * blockDim.x + threadIdx.x;
  if (k < Dm) {
    float acc = 0.f;
    for (int n = 0; n < Dm; ++n) acc += Wv[(size_t)n * Dm + k];
    wvsum[k] = acc;
  }
  if (blockIdx.x == 0 && threadIdx.x == 0) {
    float a = 0.f;
    for (int n = 0; n < Dm; ++n) a += bv[n];
    *bvsum = a;
  }
}

// ------- x: fp32 (B,S,D) -> bf16 (S,B,D); fused vsum[s,b] = x_row·wvsum + Σbv -------
__global__ __launch_bounds__(256)
void convert_x_kernel(const float* __restrict__ x,
                      const float* __restrict__ wvsum,
                      const float* __restrict__ bvsum,
                      unsigned short* __restrict__ xb,
                      float* __restrict__ vsum) {
  const int wave = threadIdx.x >> 5;
  const int lane = threadIdx.x & 31;
  const int rowIn = blockIdx.x * 8 + wave;      // rowIn = b*S + s
  const int b = rowIn >> 11;                    // / 2048
  const int s = rowIn & 2047;
  const size_t src = (size_t)rowIn * Dm;
  const size_t dst = ((size_t)s * Bm + b) * Dm;
  float dot = 0.f;
#pragma unroll
  for (int j = 0; j < 16; ++j) {
    int idx = lane + j * 32;
    float f = x[src + idx];
    dot = fmaf(f, wvsum[idx], dot);
    xb[dst + idx] = f32_bf16(f);
  }
#pragma unroll
  for (int off = 16; off; off >>= 1) dot += __shfl_xor(dot, off, 32);
  if (lane == 0) vsum[(size_t)s * Bm + b] = dot + *bvsum;
}

// ------------- y = xb @ W^T + bias, bf16 in / bf16 out, f32 accum -------------
// block: 256 thr = 8 waves; block tile M=128, N=64.
// The 64x512 bf16 W strip (64KB) is DMA'd once into LDS by the TDM (wave 0),
// then every wave reads B fragments with ds_load while A streams from global.
__global__ __launch_bounds__(256)
void proj_gemm_kernel(const unsigned short* __restrict__ xb,
                      const unsigned short* __restrict__ Wb,
                      const float* __restrict__ bias,
                      unsigned short* __restrict__ outb) {
  __shared__ unsigned short wtile[64 * Dm];     // 64 KB
  const int wave = threadIdx.x >> 5;
  const int lane = threadIdx.x & 31;
  const int laneLo = lane & 15;
  const bool hi = lane >= 16;

  const int m0 = blockIdx.x * 128 + wave * 16;
  const int n0 = blockIdx.y * 64;

  if (threadIdx.x < 32) {
    // one TDM op per block: rows n0..n0+63 of W, all 512 k
    tdm_load_2d_bf16(Wb + (size_t)n0 * Dm,
                     (unsigned)(uintptr_t)(void*)wtile,
                     /*tileD0=*/Dm, /*tileD1=*/64,
                     /*tensD0=*/Dm, /*tensD1=*/Dm - (unsigned)n0,
                     /*stride0=*/Dm);
    __builtin_amdgcn_s_wait_tensorcnt(0);
  }
  __syncthreads();

  v8f acc[4] = {};

  const __bf16* A  = reinterpret_cast<const __bf16*>(xb);
  const __bf16* Bl = reinterpret_cast<const __bf16*>(wtile);
  const int aRow = m0 + laneLo;

  for (int k0 = 0; k0 < Dm; k0 += 32) {
    // A fragment: lane<16 holds K k0..k0+15; lane>=16 holds K k0+16..k0+31
    const __bf16* ap = A + (size_t)aRow * Dm + k0 + (hi ? 16 : 0);
    __builtin_prefetch((const void*)(ap + 32), 0, 0);   // next k-step
    v8bf alo = *reinterpret_cast<const v8bf*>(ap);
    v8bf ahi = *reinterpret_cast<const v8bf*>(ap + 8);
    v16bf afrag;
#pragma unroll
    for (int i = 0; i < 8; ++i) { afrag[i] = alo[i]; afrag[i + 8] = ahi[i]; }
#pragma unroll
    for (int t = 0; t < 4; ++t) {
      // B fragment from LDS: row (t*16+laneLo) of strip, lanes<16 K k0..k0+15
      const __bf16* bp = Bl + (size_t)(t * 16 + laneLo) * Dm + k0 + (hi ? 16 : 0);
      v16bf bfrag = *reinterpret_cast<const v16bf*>(bp);
      acc[t] = __builtin_amdgcn_wmma_f32_16x16x32_bf16(
          false, afrag, false, bfrag, (short)0, acc[t], false, false);
    }
  }
  // epilogue: +bias, convert to bf16, scatter per C/D layout
#pragma unroll
  for (int t = 0; t < 4; ++t) {
    const int n = n0 + t * 16 + laneLo;
    const float bn = bias[n];
    const int rbase = m0 + (hi ? 8 : 0);
#pragma unroll
    for (int r = 0; r < 8; ++r)
      outb[(size_t)(rbase + r) * Dm + n] = f32_bf16(acc[t][r] + bn);
  }
}

// ---- per-s attention: scores = Q_s K_s^T / sqrt(D) (WMMA), softmax, out = attn·vsum ----
__global__ __launch_bounds__(256)
void attn_kernel(const unsigned short* __restrict__ qt,
                 const unsigned short* __restrict__ kt,
                 const float* __restrict__ vsum,
                 float* __restrict__ outp) {
  __shared__ float smem[64 * 64];
  __shared__ float vs[64];
  const int sIdx = blockIdx.x;
  const int wave = threadIdx.x >> 5;
  const int lane = threadIdx.x & 31;
  const int laneLo = lane & 15;
  const bool hi = lane >= 16;

  if (threadIdx.x < 64) vs[threadIdx.x] = vsum[(size_t)sIdx * 64 + threadIdx.x];

  const __bf16* Q  = reinterpret_cast<const __bf16*>(qt) + (size_t)sIdx * 64 * Dm;
  const __bf16* Kp = reinterpret_cast<const __bf16*>(kt) + (size_t)sIdx * 64 * Dm;

  // 4x4 grid of 16x16 score tiles; wave w -> row-tile w/2, col-tiles {2*(w&1), 2*(w&1)+1}
  const int bi  = wave >> 1;
  const int ci0 = (wave & 1) * 2;
  v8f acc[2] = {};
  const int aRow = bi * 16 + laneLo;

  for (int k0 = 0; k0 < Dm; k0 += 32) {
    const __bf16* ap = Q + (size_t)aRow * Dm + k0 + (hi ? 16 : 0);
    v8bf alo = *reinterpret_cast<const v8bf*>(ap);
    v8bf ahi = *reinterpret_cast<const v8bf*>(ap + 8);
    v16bf afrag;
#pragma unroll
    for (int i = 0; i < 8; ++i) { afrag[i] = alo[i]; afrag[i + 8] = ahi[i]; }
#pragma unroll
    for (int t = 0; t < 2; ++t) {
      const int c = (ci0 + t) * 16 + laneLo;     // column of K^T = row of K_s
      const __bf16* bp = Kp + (size_t)c * Dm + k0 + (hi ? 16 : 0);
      v16bf bfrag = *reinterpret_cast<const v16bf*>(bp);
      acc[t] = __builtin_amdgcn_wmma_f32_16x16x32_bf16(
          false, afrag, false, bfrag, (short)0, acc[t], false, false);
    }
  }
  const float scale = 0.04419417382415922f;   // 1/sqrt(512)
#pragma unroll
  for (int t = 0; t < 2; ++t) {
    const int c = (ci0 + t) * 16 + laneLo;
    const int rbase = bi * 16 + (hi ? 8 : 0);
#pragma unroll
    for (int r = 0; r < 8; ++r)
      smem[(rbase + r) * 64 + c] = acc[t][r] * scale;
  }
  __syncthreads();

  // each wave: 8 rows; fused softmax + dot(attn_row, vsum_row)
#pragma unroll
  for (int j = 0; j < 8; ++j) {
    const int row = wave * 8 + j;
    float v0 = smem[row * 64 + lane];
    float v1 = smem[row * 64 + 32 + lane];
    float m = fmaxf(v0, v1);
#pragma unroll
    for (int off = 16; off; off >>= 1) m = fmaxf(m, __shfl_xor(m, off, 32));
    float e0 = __expf(v0 - m);
    float e1 = __expf(v1 - m);
    float den = e0 + e1;
    float num = e0 * vs[lane] + e1 * vs[lane + 32];
#pragma unroll
    for (int off = 16; off; off >>= 1) {
      den += __shfl_xor(den, off, 32);
      num += __shfl_xor(num, off, 32);
    }
    if (lane == 0) outp[(size_t)sIdx * 64 + row] = num / den;
  }
}

extern "C" void kernel_launch(void* const* d_in, const int* in_sizes, int n_in,
                              void* d_out, int out_size, void* d_ws, size_t ws_size,
                              hipStream_t stream) {
  (void)in_sizes; (void)n_in; (void)out_size; (void)ws_size;
  const float* x  = (const float*)d_in[0];
  const float* Wq = (const float*)d_in[1];
  const float* bq = (const float*)d_in[2];
  const float* Wk = (const float*)d_in[3];
  const float* bk = (const float*)d_in[4];
  const float* Wv = (const float*)d_in[5];
  const float* bv = (const float*)d_in[6];
  float* outp = (float*)d_out;

  char* ws = (char*)d_ws;
  size_t off = 0;
  auto carve = [&](size_t bytes) {
    char* p = ws + off;
    off += (bytes + 255) & ~(size_t)255;
    return p;
  };
  const size_t rows = (size_t)Sm * Bm;                             // 131072
  unsigned short* xb    = (unsigned short*)carve(rows * Dm * 2);   // x bf16, (S,B,D)
  unsigned short* qt    = (unsigned short*)carve(rows * Dm * 2);   // q bf16, (S,B,D)
  unsigned short* ktb   = (unsigned short*)carve(rows * Dm * 2);   // k bf16, (S,B,D)
  unsigned short* Wqb   = (unsigned short*)carve((size_t)Dm * Dm * 2);
  unsigned short* Wkb   = (unsigned short*)carve((size_t)Dm * Dm * 2);
  float*          wvsum = (float*)carve(Dm * 4);
  float*          bvsum = (float*)carve(256);
  float*          vsum  = (float*)carve(rows * 4);

  convert_w_kernel<<<(Dm * Dm + 255) / 256, 256, 0, stream>>>(Wq, Wk, Wqb, Wkb);
  wv_colsum_kernel<<<(Dm + 255) / 256, 256, 0, stream>>>(Wv, bv, wvsum, bvsum);
  convert_x_kernel<<<(unsigned)(rows / 8), 256, 0, stream>>>(x, wvsum, bvsum, xb, vsum);
  proj_gemm_kernel<<<dim3((unsigned)(rows / 128), Dm / 64), 256, 0, stream>>>(xb, Wqb, bq, qt);
  proj_gemm_kernel<<<dim3((unsigned)(rows / 128), Dm / 64), 256, 0, stream>>>(xb, Wkb, bk, ktb);
  attn_kernel<<<Sm, 256, 0, stream>>>(qt, ktb, vsum, outp);
}
